// MolmoeMlpExpert_16398185136855
// MI455X (gfx1250) — matrix operand order repaired and checked by reading, not compiled, not run on hardware
//
#include <hip/hip_runtime.h>
#include <hip/hip_bf16.h>

// ---------------------------------------------------------------------------
// MoE SwiGLU MLP for MI455X (gfx1250), wave32 + V_WMMA_F32_16X16X32_BF16.
// T=2048 tokens, D=2048, H=1024, E=64 experts, K=8, C=512 capacity.
// Weight traffic (1.6 GB fp32) dominates -> bf16 WMMA with fp32 accumulate,
// 64-row x 128-col tiles per workgroup to amortize weight streaming.
// fp32->bf16 via __builtin_convertvector (legalizes to v_cvt_pk_bf16_f32
// when the target has it; RNE rounding either way).
// ---------------------------------------------------------------------------

#define DDIM 2048
#define HDIM 1024
#define EEXP 64
#define TOPK 8
#define CAP  512
#define TTOK 2048
#define LDK  40   // padded LDS k-stride (elements) to avoid bank conflicts

typedef __attribute__((ext_vector_type(16))) __bf16 v16bf;
typedef __attribute__((ext_vector_type(2)))  __bf16 v2bf;
typedef __attribute__((ext_vector_type(2)))  float  v2f;
typedef __attribute__((ext_vector_type(8)))  float  v8f;

union AFrag { v16bf v; unsigned short s[16]; unsigned int u[8]; };
union CFrag { v8f   v; float f[8]; };

// Packed fp32x2 -> bf16x2 (vector fptrunc; backend may select v_cvt_pk_bf16_f32).
__device__ __forceinline__ unsigned int f2bf2(float lo, float hi) {
    v2f s;
    s[0] = lo;
    s[1] = hi;
    union { v2bf v; unsigned int u; } r;
    r.v = __builtin_convertvector(s, v2bf);
    return r.u;
}

#define WMMA_BF16(a, b, c) \
    __builtin_amdgcn_wmma_f32_16x16x32_bf16(false, (a), false, (b), (short)0, (c), false, false)

// A fragment: 16x32 bf16 tile, row-major in LDS (stride lda elems).
// ISA layout: lanes 0-15 M=lane, kbase=0; lanes 16-31 M=lane-16, kbase=8.
// VGPR p<4: K=kb+2p,+1 ; p>=4: K=kb+16+2(p-4),+1  -> contiguous pairs -> dword reads.
__device__ __forceinline__ v16bf lds_a_frag(const unsigned short* A, int lda, int lane) {
    AFrag a;
    int M  = lane & 15;
    int kb = (lane < 16) ? 0 : 8;
    const unsigned short* row = A + M * lda;
#pragma unroll
    for (int p = 0; p < 8; ++p) {
        int k = kb + 2 * p + ((p >= 4) ? 8 : 0);
        a.u[p] = *(const unsigned int*)(row + k);
    }
    return a.v;
}

// B fragment: 32x16 bf16 (KxN), staged TRANSPOSED in LDS as BT[col][k] (stride ldk).
// Lanes 0-15: N=lane, kbase=0; lanes 16-31: N=lane-16, kbase=8; element i<8 <-> K=kb+i, i>=8 <-> K=kb+16+(i-8).
__device__ __forceinline__ v16bf lds_bt_frag(const unsigned short* BT, int ldk, int lane) {
    AFrag b;
    int N  = lane & 15;
    int kb = (lane < 16) ? 0 : 8;
    const unsigned short* col = BT + N * ldk + kb;
#pragma unroll
    for (int p = 0; p < 8; ++p)
        b.u[p] = *(const unsigned int*)(col + 2 * p + ((p >= 4) ? 8 : 0));
    return b.v;
}

// Store float4 as 4 packed bf16 (8B aligned) -> single ds_store_b64.
__device__ __forceinline__ void st_bf16x4(unsigned short* dst, float4 f) {
    uint2 v;
    v.x = f2bf2(f.x, f.y);
    v.y = f2bf2(f.z, f.w);
    *(uint2*)dst = v;
}

// Transposed scatter of float4 into BT[col][k] (one packed cvt per 2 elems;
// hi halves store via ds_store_b16_d16_hi).
__device__ __forceinline__ void st_bt_x4(unsigned short* bt, int cb, int k, float4 f) {
    unsigned int xy = f2bf2(f.x, f.y);
    unsigned int zw = f2bf2(f.z, f.w);
    bt[(cb + 0) * LDK + k] = (unsigned short)xy;
    bt[(cb + 1) * LDK + k] = (unsigned short)(xy >> 16);
    bt[(cb + 2) * LDK + k] = (unsigned short)zw;
    bt[(cb + 3) * LDK + k] = (unsigned short)(zw >> 16);
}

// ------------------------- router: logits = x @ gate_w ---------------------
// grid 128 blocks x 128 threads (4 waves); block = 16 tokens x 64 experts.
__global__ __launch_bounds__(128) void router_kernel(const float* __restrict__ x,
                                                     const float* __restrict__ gw,
                                                     float* __restrict__ logits) {
    __shared__ unsigned short sA[16 * LDK];
    __shared__ unsigned short sBT[64 * LDK];
    int tid = threadIdx.x, lane = tid & 31, w = tid >> 5;
    int t0 = blockIdx.x * 16;

    CFrag acc;
#pragma unroll
    for (int i = 0; i < 8; ++i) acc.f[i] = 0.f;

    for (int d0 = 0; d0 < DDIM; d0 += 32) {
        __syncthreads();
        {   // A: 16 rows x 32 = 128 float4, one per thread
            int row = tid >> 3, q = tid & 7;
            float4 f = *(const float4*)(x + (size_t)(t0 + row) * DDIM + d0 + q * 4);
            st_bf16x4(sA + row * LDK + q * 4, f);
        }
#pragma unroll
        for (int p = 0; p < 4; ++p) {   // B: 32 x 64 -> transposed
            int idx = p * 128 + tid;
            int k = idx >> 4, c4 = idx & 15;
            float4 f = *(const float4*)(gw + (size_t)(d0 + k) * EEXP + c4 * 4);
            st_bt_x4(sBT, c4 * 4, k, f);
        }
        __syncthreads();
        v16bf a = lds_a_frag(sA, LDK, lane);
        v16bf b = lds_bt_frag(sBT + (w * 16) * LDK, LDK, lane);
        acc.v = WMMA_BF16(a, b, acc.v);
    }
    int N = lane & 15, mh = (lane < 16) ? 0 : 8;
#pragma unroll
    for (int r = 0; r < 8; ++r)
        logits[(size_t)(t0 + mh + r) * EEXP + w * 16 + N] = acc.f[r];
}

// ---------------------- softmax + top-8 per token --------------------------
__global__ __launch_bounds__(256) void topk_kernel(const float* __restrict__ logits,
                                                   float* __restrict__ wv,
                                                   int* __restrict__ sel) {
    int t = blockIdx.x * 256 + threadIdx.x;
    if (t >= TTOK) return;
    float l[EEXP];
    float mx = -3.0e38f;
#pragma unroll 8
    for (int i = 0; i < EEXP; ++i) { l[i] = logits[(size_t)t * EEXP + i]; mx = fmaxf(mx, l[i]); }
    float sum = 0.f;
#pragma unroll 8
    for (int i = 0; i < EEXP; ++i) { l[i] = __expf(l[i] - mx); sum += l[i]; }
    float inv = 1.f / sum;
#pragma unroll
    for (int j = 0; j < TOPK; ++j) {
        int   bi = 0;
        float bv = l[0];
#pragma unroll 8
        for (int i = 1; i < EEXP; ++i) { if (l[i] > bv) { bv = l[i]; bi = i; } }
        sel[t * TOPK + j] = bi;
        wv[t * TOPK + j]  = bv * inv;
        l[bi] = -1.f;
    }
}

// ------------- deterministic dispatch (matches reference cumsum order) -----
__global__ void assign_kernel(const int* __restrict__ sel, const float* __restrict__ wv,
                              int* __restrict__ idxb, float* __restrict__ wtb,
                              int* __restrict__ cnt) {
    int e = threadIdx.x;
    if (e >= EEXP) return;
    int c = 0;
    const int4* s4 = (const int4*)sel;
    for (int q = 0; q < (TTOK * TOPK) / 4; ++q) {
        int4 v = s4[q];
        int p = q * 4;
        if (v.x == e && c < CAP) { idxb[e * CAP + c] = (p + 0) >> 3; wtb[e * CAP + c] = wv[p + 0]; ++c; }
        if (v.y == e && c < CAP) { idxb[e * CAP + c] = (p + 1) >> 3; wtb[e * CAP + c] = wv[p + 1]; ++c; }
        if (v.z == e && c < CAP) { idxb[e * CAP + c] = (p + 2) >> 3; wtb[e * CAP + c] = wv[p + 2]; ++c; }
        if (v.w == e && c < CAP) { idxb[e * CAP + c] = (p + 3) >> 3; wtb[e * CAP + c] = wv[p + 3]; ++c; }
    }
    cnt[e] = c;
}

// --------- gemm1: h = silu(x@Wg) * (x@Wu), bf16 out, gathered rows ---------
// grid (E*8, H/128), block 256 (8 waves). Block = 64 rows x 128 cols.
// Wave w owns col-tile w (16 cols) and loops 4 row-tiles (B-frag reuse).
__global__ __launch_bounds__(256) void gemm1_kernel(const float* __restrict__ x,
                                                    const float* __restrict__ wg,
                                                    const float* __restrict__ wu,
                                                    const int* __restrict__ idxb,
                                                    const int* __restrict__ cnt,
                                                    unsigned short* __restrict__ hbuf) {
    __shared__ unsigned short sA[64 * LDK];
    __shared__ unsigned short sGT[128 * LDK];
    __shared__ unsigned short sUT[128 * LDK];
    __shared__ int tok[64];

    int tid = threadIdx.x, lane = tid & 31, w = tid >> 5;
    int e  = blockIdx.x >> 3;
    int rb = (blockIdx.x & 7) * 64;
    int ce = cnt[e];
    if (rb >= ce) return;                        // whole tile empty: skip
    int h0 = blockIdx.y * 128;

    if (tid < 64) {
        int rg = rb + tid;
        tok[tid] = (rg < ce) ? idxb[e * CAP + rg] : -1;
    }

    CFrag accG[4], accU[4];
#pragma unroll
    for (int rt = 0; rt < 4; ++rt)
#pragma unroll
        for (int i = 0; i < 8; ++i) { accG[rt].f[i] = 0.f; accU[rt].f[i] = 0.f; }

    const float* wge = wg + (size_t)e * DDIM * HDIM;
    const float* wue = wu + (size_t)e * DDIM * HDIM;

    for (int d0 = 0; d0 < DDIM; d0 += 32) {
        __syncthreads();
#pragma unroll
        for (int p = 0; p < 2; ++p) {            // A: 64 x 32 gathered from x
            int idx = p * 256 + tid;
            int row = idx >> 3, q = idx & 7;
            int t = tok[row];
            float4 f = make_float4(0.f, 0.f, 0.f, 0.f);
            if (t >= 0) f = *(const float4*)(x + (size_t)t * DDIM + d0 + q * 4);
            st_bf16x4(sA + row * LDK + q * 4, f);
        }
#pragma unroll
        for (int p = 0; p < 4; ++p) {            // B gate+up: 32 x 128 each, transposed
            int idx = p * 256 + tid;
            int k = idx >> 5, c4 = idx & 31;
            size_t off = (size_t)(d0 + k) * HDIM + h0 + c4 * 4;
            float4 g4 = *(const float4*)(wge + off);
            float4 u4 = *(const float4*)(wue + off);
            if (d0 + 32 < DDIM) {                // stream hint -> global_prefetch_b8
                __builtin_prefetch(wge + off + 32 * HDIM, 0, 1);
                __builtin_prefetch(wue + off + 32 * HDIM, 0, 1);
            }
            st_bt_x4(sGT, c4 * 4, k, g4);
            st_bt_x4(sUT, c4 * 4, k, u4);
        }
        __syncthreads();
        v16bf bg = lds_bt_frag(sGT + (w * 16) * LDK, LDK, lane);
        v16bf bu = lds_bt_frag(sUT + (w * 16) * LDK, LDK, lane);
#pragma unroll
        for (int rt = 0; rt < 4; ++rt) {
            v16bf a = lds_a_frag(sA + rt * 16 * LDK, LDK, lane);
            accG[rt].v = WMMA_BF16(a, bg, accG[rt].v);
            accU[rt].v = WMMA_BF16(a, bu, accU[rt].v);
        }
    }

    int N = lane & 15, mh = (lane < 16) ? 0 : 8;
#pragma unroll
    for (int rt = 0; rt < 4; ++rt)
#pragma unroll
        for (int r = 0; r < 8; r += 2) {         // pairs -> one packed cvt per 2
            int row0 = rt * 16 + mh + r;
            float g0 = accG[rt].f[r],     u0 = accU[rt].f[r];
            float g1 = accG[rt].f[r + 1], u1 = accU[rt].f[r + 1];
            float h0v = g0 * (1.f / (1.f + __expf(-g0))) * u0;   // silu(g)*u
            float h1v = g1 * (1.f / (1.f + __expf(-g1))) * u1;
            unsigned int pk = f2bf2(h0v, h1v);
            size_t base = ((size_t)e * CAP + rb + row0) * HDIM + h0 + w * 16 + N;
            hbuf[base]        = (unsigned short)pk;
            hbuf[base + HDIM] = (unsigned short)(pk >> 16);
        }
}

// ------------- gemm2: y = h @ Wd, weighted atomic combine into out ---------
// grid (E*8, D/128), block 256 (8 waves).
__global__ __launch_bounds__(256) void gemm2_kernel(const unsigned short* __restrict__ hbuf,
                                                    const float* __restrict__ wd,
                                                    const int* __restrict__ idxb,
                                                    const float* __restrict__ wtb,
                                                    const int* __restrict__ cnt,
                                                    float* __restrict__ out) {
    __shared__ unsigned short sA[64 * LDK];
    __shared__ unsigned short sBT[128 * LDK];
    __shared__ int   tok[64];
    __shared__ float wrow[64];

    int tid = threadIdx.x, lane = tid & 31, w = tid >> 5;
    int e  = blockIdx.x >> 3;
    int rb = (blockIdx.x & 7) * 64;
    int ce = cnt[e];
    if (rb >= ce) return;
    int d0 = blockIdx.y * 128;

    if (tid < 64) {
        int rg = rb + tid;
        int valid = rg < ce;
        tok[tid]  = valid ? idxb[e * CAP + rg] : -1;
        wrow[tid] = valid ? wtb[e * CAP + rg] : 0.f;
    }

    CFrag acc[4];
#pragma unroll
    for (int rt = 0; rt < 4; ++rt)
#pragma unroll
        for (int i = 0; i < 8; ++i) acc[rt].f[i] = 0.f;

    const float* wde = wd + (size_t)e * HDIM * DDIM;

    for (int k0 = 0; k0 < HDIM; k0 += 32) {
        __syncthreads();
#pragma unroll
        for (int p = 0; p < 4; ++p) {            // A: 64 x 32 bf16 direct from hbuf
            int idx = p * 256 + tid;
            int row = idx >> 4, dw = idx & 15;
            unsigned int v = *(const unsigned int*)(hbuf + ((size_t)e * CAP + rb + row) * HDIM + k0 + dw * 2);
            *(unsigned int*)(sA + row * LDK + dw * 2) = v;
        }
#pragma unroll
        for (int p = 0; p < 4; ++p) {            // B: 32 x 128 fp32 -> bf16 transposed
            int idx = p * 256 + tid;
            int k = idx >> 5, c4 = idx & 31;
            size_t off = (size_t)(k0 + k) * DDIM + d0 + c4 * 4;
            float4 f = *(const float4*)(wde + off);
            if (k0 + 32 < HDIM) __builtin_prefetch(wde + off + 32 * DDIM, 0, 1);
            st_bt_x4(sBT, c4 * 4, k, f);
        }
        __syncthreads();
        v16bf b = lds_bt_frag(sBT + (w * 16) * LDK, LDK, lane);
#pragma unroll
        for (int rt = 0; rt < 4; ++rt) {
            v16bf a = lds_a_frag(sA + rt * 16 * LDK, LDK, lane);
            acc[rt].v = WMMA_BF16(a, b, acc[rt].v);
        }
    }

    int N = lane & 15, mh = (lane < 16) ? 0 : 8;
#pragma unroll
    for (int rt = 0; rt < 4; ++rt)
#pragma unroll
        for (int r = 0; r < 8; ++r) {
            int row = rt * 16 + mh + r;
            int t = tok[row];
            if (t >= 0) {
                float val = acc[rt].f[r] * wrow[row];
                __hip_atomic_fetch_add(out + (size_t)t * DDIM + d0 + w * 16 + N, val,
                                       __ATOMIC_RELAXED, __HIP_MEMORY_SCOPE_AGENT);
            }
        }
}

// ---------------------------------------------------------------------------
extern "C" void kernel_launch(void* const* d_in, const int* in_sizes, int n_in,
                              void* d_out, int out_size, void* d_ws, size_t ws_size,
                              hipStream_t stream) {
    const float* x  = (const float*)d_in[0];   // [T, D]
    const float* gw = (const float*)d_in[1];   // [D, E]
    const float* wg = (const float*)d_in[2];   // [E, D, H]
    const float* wu = (const float*)d_in[3];   // [E, D, H]
    const float* wd = (const float*)d_in[4];   // [E, H, D]

    float* out    = (float*)d_out;                       // [T, D]
    float* logits = out + (size_t)TTOK * DDIM;           // [T, E] (second output)

    // workspace carve (256B aligned): wv, sel, idx, wt, cnt, hbuf(bf16)
    char* ws = (char*)d_ws;
    float*          wv   = (float*)(ws + 0);                     // 64 KB
    int*            selb = (int*)  (ws + 65536);                 // 64 KB
    int*            idxb = (int*)  (ws + 131072);                // 128 KB
    float*          wtb  = (float*)(ws + 262144);                // 128 KB
    int*            cntb = (int*)  (ws + 393216);                // 256 B
    unsigned short* hbuf = (unsigned short*)(ws + 393472);       // 64 MB

    // out region is accumulated via atomics -> must be zeroed every call
    hipMemsetAsync(out, 0, (size_t)TTOK * DDIM * sizeof(float), stream);

    router_kernel<<<TTOK / 16, 128, 0, stream>>>(x, gw, logits);
    topk_kernel<<<TTOK / 256, 256, 0, stream>>>(logits, wv, selb);
    assign_kernel<<<1, 64, 0, stream>>>(selb, wv, idxb, wtb, cntb);
    gemm1_kernel<<<dim3(EEXP * 8, HDIM / 128), 256, 0, stream>>>(x, wg, wu, idxb, cntb, hbuf);
    gemm2_kernel<<<dim3(EEXP * 8, DDIM / 128), 256, 0, stream>>>(hbuf, wd, idxb, wtb, cntb, out);
}
